// BidirectionalLayerFeatCosine_78426102825594
// MI455X (gfx1250) — matrix operand order repaired and checked by reading, not compiled, not run on hardware
//
#include <hip/hip_runtime.h>
#include <hip/hip_bf16.h>

// ---------------------------------------------------------------------------
// BidirectionalLayerFeatCosine for MI455X (gfx1250, wave32, WMMA bf16 + TDM)
//
// GEMM operands pre-packed to bf16 "k-pair" dwords -> inner loop is
// global_load_b128 (A) + ds_load (B) + v_wmma_f32_16x16x32_bf16, no cvt VALU.
// X tiles are streamed into double-buffered LDS by the Tensor Data Mover:
// wave0 issues TDM(tile s+1) then waits tensorcnt<=1 (in-order completion
// guarantees tile s landed), overlapping DMA with WMMA compute.
// ---------------------------------------------------------------------------

typedef __bf16 bf16_t;
typedef bf16_t v16bf __attribute__((ext_vector_type(16)));
typedef float  v8f   __attribute__((ext_vector_type(8)));
typedef int    v8i   __attribute__((ext_vector_type(8)));
typedef unsigned int u32x4 __attribute__((ext_vector_type(4)));
typedef int    i32x4 __attribute__((ext_vector_type(4)));
typedef int    i32x8 __attribute__((ext_vector_type(8)));

static constexpr int B   = 2;
static constexpr int N   = 4096;
static constexpr int C   = 256;
static constexpr int M0  = 256;
static constexpr int M2  = 512;
static constexpr int NK  = 16;    // NSAMPLE
static constexpr float EPS = 1e-5f;
static constexpr float NEG_SLOPE = 0.1f;

#ifdef __has_builtin
# if __has_builtin(__builtin_amdgcn_tensor_load_to_lds) && __has_builtin(__builtin_amdgcn_s_wait_tensorcnt)
#  define HAVE_TDM 1
# endif
#endif
#ifndef HAVE_TDM
# define HAVE_TDM 0
#endif

__device__ __forceinline__ bf16_t f2bf(float f) {
  unsigned u = __builtin_bit_cast(unsigned, f);
  unsigned r = u + 0x7FFFu + ((u >> 16) & 1u);   // round-to-nearest-even
  unsigned short h = (unsigned short)(r >> 16);
  return __builtin_bit_cast(bf16_t, h);
}
__device__ __forceinline__ unsigned pack2bf(float lo, float hi) {
  unsigned a = (unsigned)__builtin_bit_cast(unsigned short, f2bf(lo));
  unsigned b = (unsigned)__builtin_bit_cast(unsigned short, f2bf(hi));
  return a | (b << 16);
}

#if HAVE_TDM
// Issue a TDM load of a 16(c2-rows) x 32(m-cols) dword tile into LDS.
// D# per ISA 8.3/8.4: group0 = {count=1, lds_addr, global_addr, type=2},
// group1 = {data_size=4B, tensor dims (M x C2), tile 32x16, dim0_stride=M}.
__device__ __forceinline__ void tdm_tile_load(const unsigned* gsrc, unsigned lds_off,
                                              int M, int C2)
{
  unsigned long long ga = (unsigned long long)(uintptr_t)gsrc;
  u32x4 g0 = { 1u,
               lds_off,
               (unsigned)(ga & 0xffffffffull),
               (unsigned)((ga >> 32) & 0x01ffffffull) | (2u << 30) };
  i32x8 g1 = { (int)(2u << 16),                                       // data_size=4B
               (int)(((unsigned)M & 0xffffu) << 16),                  // tensor_dim0 lo
               (int)((((unsigned)M >> 16) & 0xffffu) |
                     (((unsigned)C2 & 0xffffu) << 16)),               // dim0 hi | dim1 lo
               (int)((((unsigned)C2 >> 16) & 0xffffu) | (32u << 16)), // dim1 hi | tile_dim0=32
               (int)16u,                                              // tile_dim1=16, tile_dim2=0
               (int)(unsigned)M,                                      // dim0_stride[31:0]
               0, 0 };
  i32x4 gz = { 0, 0, 0, 0 };
#if __clang_major__ >= 23
  i32x8 g4 = { 0, 0, 0, 0, 0, 0, 0, 0 };
  __builtin_amdgcn_tensor_load_to_lds(g0, g1, gz, gz, g4, 0);
#else
  __builtin_amdgcn_tensor_load_to_lds(g0, g1, gz, gz, 0);
#endif
}
#endif

// ---------------------------------------------------------------------------
// GEMM: out[b][o][m] = sum_c W[o][c]*X[b][c][m] + bias[o]
// Wpk: [O][C/2] dwords (bf16 pair).  Xpk: [B][C/2][M] dwords.
// Block = 128 thr (4 waves) -> 32(O) x 32(M) output block; wave w computes
// O-tile (w>>1), M-half (w&1).  X tiles double-buffered in LDS via TDM.
// ---------------------------------------------------------------------------
__global__ void __launch_bounds__(128)
gemm_wmma_bf16pk(const unsigned* __restrict__ Wpk,
                 const unsigned* __restrict__ Xpk,
                 const float* __restrict__ bias,
                 float* __restrict__ out,
                 int O, int Cdim, int M)
{
  const int C2 = Cdim >> 1;
  const int mb = blockIdx.x * 32;
  const int ob = blockIdx.y * 32;
  const int b  = blockIdx.z;
  const unsigned* Xb = Xpk + (size_t)b * C2 * M;
  float* Ob = out + (size_t)b * O * M;

  const int tid  = threadIdx.x;
  const int wave = tid >> 5;
  const int lane = tid & 31;
  const int half = lane >> 4;
  const int l15  = lane & 15;
  const int otile = ob + (wave >> 1) * 16;
  const int mloc  = (wave & 1) * 16;

  __shared__ unsigned ldsX[2][16 * 32];   // ping-pong: 16 c2-rows x 32 m-cols

#if HAVE_TDM
  if (wave == 0)
    tdm_tile_load(Xb + /*c2base=*/0 * M + mb,
                  (unsigned)(uintptr_t)&ldsX[0][0], M, C2);
#endif

  v8f acc = {};
  int buf = 0;
  for (int k0 = 0; k0 < Cdim; k0 += 32, buf ^= 1) {
    const int c2base = k0 >> 1;
#if HAVE_TDM
    if (wave == 0) {
      if (k0 + 32 < Cdim) {
        // overlap: issue next tile DMA, then wait only until current landed
        tdm_tile_load(Xb + (size_t)(c2base + 16) * M + mb,
                      (unsigned)(uintptr_t)&ldsX[buf ^ 1][0], M, C2);
        __builtin_amdgcn_s_wait_tensorcnt(1);
      } else {
        __builtin_amdgcn_s_wait_tensorcnt(0);
      }
    }
#else
    for (int i = tid; i < 16 * 32; i += 128)
      ldsX[buf][i] = Xb[(size_t)(c2base + (i >> 5)) * M + mb + (i & 31)];
#endif
    __syncthreads();                     // current tile visible to all waves

    if (k0 + 32 < Cdim)
      __builtin_prefetch(Wpk + (size_t)(otile + l15) * C2 + (c2base + 16), 0, 0);

    v8i ai, bi;
    #pragma unroll
    for (int j = 0; j < 8; ++j) {
      // A 16x32 bf16 layout: pair index covers ISA K mapping (even K pairs)
      const int c2off = (j < 4) ? (j + 4 * half) : (8 + (j - 4) + 4 * half);
      ai[j] = (int)Wpk[(size_t)(otile + l15) * C2 + c2base + c2off];
      // B 32x16 bf16 layout: K pair = c2off (same as A), N = l15
      bi[j] = (int)ldsX[buf][c2off * 32 + mloc + l15];
    }
    acc = __builtin_amdgcn_wmma_f32_16x16x32_bf16(
        false, __builtin_bit_cast(v16bf, ai),
        false, __builtin_bit_cast(v16bf, bi),
        (short)0, acc, false, false);
    __syncthreads();                     // release buf for the in-flight DMA
  }
  #pragma unroll
  for (int v = 0; v < 8; ++v) {
    const int row = otile + v + 8 * half;       // C/D: M = v + 8*(lane>>4)
    Ob[(size_t)row * M + mb + mloc + l15] = acc[v] + bias[row];
  }
}

// ---------------------------------------------------------------------------
// Pack fp32 weight [O][C] -> bf16-pair dwords [O][C/2]
// ---------------------------------------------------------------------------
__global__ void __launch_bounds__(256)
pack_w_kernel(const float* __restrict__ W, unsigned* __restrict__ Wpk, int O, int Cdim)
{
  const int C2 = Cdim >> 1;
  int i = blockIdx.x * blockDim.x + threadIdx.x;
  if (i >= O * C2) return;
  int o = i / C2, c2 = i % C2;
  Wpk[i] = pack2bf(W[(size_t)o * Cdim + 2 * c2], W[(size_t)o * Cdim + 2 * c2 + 1]);
}

// ---------------------------------------------------------------------------
// Pack fp32 activations [B][Cdim][M] -> bf16-pair dwords [B][Cdim/2][M]
// ---------------------------------------------------------------------------
__global__ void __launch_bounds__(256)
pack_x_kernel(const float* __restrict__ X, unsigned* __restrict__ Xpk, int Cdim, int M)
{
  const int C2 = Cdim >> 1;
  size_t i = (size_t)blockIdx.x * blockDim.x + threadIdx.x;
  if (i >= (size_t)B * C2 * M) return;
  int m = (int)(i % (size_t)M);
  size_t r = i / (size_t)M;
  int c2 = (int)(r % (size_t)C2);
  int b  = (int)(r / (size_t)C2);
  const float* p = X + ((size_t)b * Cdim + 2 * c2) * M + m;
  Xpk[i] = pack2bf(p[0], p[(size_t)M]);
}

// ---------------------------------------------------------------------------
// Normalize descriptors along C; store bf16-pair dwords [B][C/2][N]
// ---------------------------------------------------------------------------
__global__ void __launch_bounds__(256)
normalize_pk(const float* __restrict__ knn, unsigned* __restrict__ outn)
{
  int t = blockIdx.x * blockDim.x + threadIdx.x;
  if (t >= B * N) return;
  int b = t / N, n = t % N;
  const float* p = knn + (size_t)b * C * N + n;
  float ss = 0.f;
  for (int c = 0; c < C; ++c) { float v = p[(size_t)c * N]; ss += v * v; }
  float scale = 1.0f / fmaxf(sqrtf(ss), 1e-8f);
  unsigned* q = outn + (size_t)b * (C / 2) * N + n;
  for (int c2 = 0; c2 < C / 2; ++c2)
    q[(size_t)c2 * N] = pack2bf(p[(size_t)(2 * c2) * N] * scale,
                                p[(size_t)(2 * c2 + 1) * N] * scale);
}

// ---------------------------------------------------------------------------
// Fused cosine+euclid distances with streaming top-8/top-8 (one wave / 16 rows)
// ---------------------------------------------------------------------------
__global__ void __launch_bounds__(32)
topk_kernel(const unsigned* __restrict__ fa, const unsigned* __restrict__ fb,
            const float* __restrict__ xyzA, const float* __restrict__ xyzB,
            int* __restrict__ knn_idx)
{
  __shared__ float s_dot[16][17];
  __shared__ float s_xyz[3][16];
  const int n1 = blockIdx.x * 16;
  const int b  = blockIdx.y;
  const int lane = threadIdx.x;
  const int half = lane >> 4;
  const int l15  = lane & 15;
  const int C2   = C / 2;

  const unsigned* fab = fa + (size_t)b * C2 * N;
  const unsigned* fbb = fb + (size_t)b * C2 * N;

  v8i afr[8];
  #pragma unroll
  for (int s = 0; s < 8; ++s) {
    const int c2b = s * 16;
    #pragma unroll
    for (int j = 0; j < 8; ++j) {
      const int c2off = (j < 4) ? (j + 4 * half) : (8 + (j - 4) + 4 * half);
      afr[s][j] = (int)fab[(size_t)(c2b + c2off) * N + n1 + l15];
    }
  }

  float rx = 0.f, ry = 0.f, rz = 0.f;
  if (lane < 16) {
    const float* xa = xyzA + (size_t)b * 3 * N;
    rx = xa[n1 + lane]; ry = xa[N + n1 + lane]; rz = xa[2 * N + n1 + lane];
  }

  float bp[8];  int bpi[8];
  float bfv[8]; int bfi[8];
  #pragma unroll
  for (int i = 0; i < 8; ++i) { bp[i] = 1e30f; bpi[i] = 0; bfv[i] = -1e30f; bfi[i] = 0; }

  for (int n2 = 0; n2 < N; n2 += 16) {
    v8f acc = {};
    #pragma unroll
    for (int s = 0; s < 8; ++s) {
      const int c2b = s * 16;
      v8i bi;
      #pragma unroll
      for (int j = 0; j < 8; ++j)
        bi[j] = (int)fbb[(size_t)(c2b + ((j < 4) ? (j + 4 * half) : (8 + (j - 4) + 4 * half))) * N + n2 + l15];
      acc = __builtin_amdgcn_wmma_f32_16x16x32_bf16(
          false, __builtin_bit_cast(v16bf, afr[s]),
          false, __builtin_bit_cast(v16bf, bi),
          (short)0, acc, false, false);
    }
    #pragma unroll
    for (int v = 0; v < 8; ++v) s_dot[v + 8 * half][l15] = acc[v];
    if (lane < 16) {
      const float* xb = xyzB + (size_t)b * 3 * N;
      s_xyz[0][lane] = xb[n2 + lane];
      s_xyz[1][lane] = xb[N + n2 + lane];
      s_xyz[2][lane] = xb[2 * N + n2 + lane];
    }
    __syncthreads();
    if (lane < 16) {
      #pragma unroll 4
      for (int j = 0; j < 16; ++j) {
        const int gid = n2 + j;
        float dx = s_xyz[0][j] - rx, dy = s_xyz[1][j] - ry, dz = s_xyz[2][j] - rz;
        float pd = sqrtf(dx * dx + dy * dy + dz * dz);
        if (pd < bp[7]) {
          bp[7] = pd; bpi[7] = gid;
          #pragma unroll
          for (int q = 7; q > 0; --q)
            if (bp[q] < bp[q - 1]) {
              float tv = bp[q]; bp[q] = bp[q - 1]; bp[q - 1] = tv;
              int ti = bpi[q]; bpi[q] = bpi[q - 1]; bpi[q - 1] = ti;
            }
        }
        float fd = s_dot[lane][j];
        if (fd > bfv[7]) {
          bfv[7] = fd; bfi[7] = gid;
          #pragma unroll
          for (int q = 7; q > 0; --q)
            if (bfv[q] > bfv[q - 1]) {
              float tv = bfv[q]; bfv[q] = bfv[q - 1]; bfv[q - 1] = tv;
              int ti = bfi[q]; bfi[q] = bfi[q - 1]; bfi[q - 1] = ti;
            }
        }
      }
    }
    __syncthreads();
  }

  if (lane < 16) {
    int* op = knn_idx + ((size_t)b * N + n1 + lane) * NK;
    #pragma unroll
    for (int i = 0; i < 8; ++i) { op[i] = bpi[i]; op[8 + i] = bfi[i]; }
  }
}

// ---------------------------------------------------------------------------
// Stage 0 (gather + pos-conv + GN0 sums), GN stats, GN apply (+bf16 pack), max
// ---------------------------------------------------------------------------
__global__ void __launch_bounds__(256)
stage0_kernel(const float* __restrict__ pa, const float* __restrict__ pb,
              const int* __restrict__ knn_idx,
              const float* __restrict__ xyzA, const float* __restrict__ xyzB,
              const float* __restrict__ pos_w, const float* __restrict__ pos_b,
              float* __restrict__ xout, float* __restrict__ stats)
{
  const int ntile = blockIdx.x, g = blockIdx.y, b = blockIdx.z;
  const int t = threadIdx.x;
  const int cl = t >> 4, kk = t & 15;
  const int o = g * 16 + cl;
  const float w0 = pos_w[o * 3 + 0], w1 = pos_w[o * 3 + 1], w2 = pos_w[o * 3 + 2];
  const float pbias = pos_b[o];
  const float* pao = pa + ((size_t)b * M0 + o) * N;
  const float* pbo = pb + ((size_t)b * M0 + o) * N;
  const float* xA = xyzA + (size_t)b * 3 * N;
  const float* xB = xyzB + (size_t)b * 3 * N;

  float s = 0.f, s2 = 0.f;
  for (int ni = 0; ni < 16; ++ni) {
    const int n = ntile * 16 + ni;
    const int idx = knn_idx[((size_t)b * N + n) * NK + kk];
    float d0 = xB[idx] - xA[n];
    float d1 = xB[N + idx] - xA[N + n];
    float d2 = xB[2 * N + idx] - xA[2 * N + n];
    float val = pbo[idx] + pao[n] + w0 * d0 + w1 * d1 + w2 * d2 + pbias;
    xout[(((size_t)b * M0 + o) * N + n) * NK + kk] = val;
    s += val; s2 += val * val;
  }
  __shared__ float rs[256], rq[256];
  rs[t] = s; rq[t] = s2;
  __syncthreads();
  for (int st = 128; st > 0; st >>= 1) {
    if (t < st) { rs[t] += rs[t + st]; rq[t] += rq[t + st]; }
    __syncthreads();
  }
  if (t == 0) {
    atomicAdd(&stats[(b * 16 + g) * 2 + 0], rs[0]);
    atomicAdd(&stats[(b * 16 + g) * 2 + 1], rq[0]);
  }
}

__global__ void __launch_bounds__(256)
gn_stats_kernel(const float* __restrict__ x, float* __restrict__ stats,
                int Cdim, int M, int G)
{
  const int part = blockIdx.x, g = blockIdx.y, b = blockIdx.z;
  const size_t base = ((size_t)b * Cdim + (size_t)g * 16) * (size_t)M;
  const size_t total = (size_t)16 * M;
  const size_t chunk = (total + gridDim.x - 1) / gridDim.x;
  size_t lo = (size_t)part * chunk;
  size_t hi = lo + chunk; if (hi > total) hi = total;
  float s = 0.f, s2 = 0.f;
  for (size_t i = lo + threadIdx.x; i < hi; i += 256) {
    float v = x[base + i]; s += v; s2 += v * v;
  }
  __shared__ float rs[256], rq[256];
  const int t = threadIdx.x;
  rs[t] = s; rq[t] = s2;
  __syncthreads();
  for (int st = 128; st > 0; st >>= 1) {
    if (t < st) { rs[t] += rs[t + st]; rq[t] += rq[t + st]; }
    __syncthreads();
  }
  if (t == 0) {
    atomicAdd(&stats[(b * G + g) * 2 + 0], rs[0]);
    atomicAdd(&stats[(b * G + g) * 2 + 1], rq[0]);
  }
}

// GN apply + leaky. pk != nullptr: write bf16-pair-packed tensor (conv input);
// else write f32 in place (final stage before maxpool).
__global__ void __launch_bounds__(256)
gn_apply_leaky(float* __restrict__ x, const float* __restrict__ stats,
               const float* __restrict__ gw, const float* __restrict__ gb,
               unsigned short* __restrict__ pk,
               int Cdim, int M, int G, float invcnt)
{
  size_t i = (size_t)blockIdx.x * blockDim.x + threadIdx.x;
  const size_t tot = (size_t)B * Cdim * M;
  if (i >= tot) return;
  const int m = (int)(i % (size_t)M);
  const int c = (int)((i / (size_t)M) % (size_t)Cdim);
  const int b = (int)(i / ((size_t)Cdim * M));
  const int g = c >> 4;
  const float s  = stats[(b * G + g) * 2 + 0];
  const float s2 = stats[(b * G + g) * 2 + 1];
  const float mu = s * invcnt;
  const float var = fmaxf(s2 * invcnt - mu * mu, 0.f);
  const float r = rsqrtf(var + EPS);
  float v = (x[i] - mu) * r * gw[c] + gb[c];
  v = v >= 0.f ? v : NEG_SLOPE * v;
  if (pk) {
    const size_t C2 = (size_t)(Cdim >> 1);
    pk[(((size_t)b * C2 + (size_t)(c >> 1)) * M + m) * 2 + (c & 1)] =
        __builtin_bit_cast(unsigned short, f2bf(v));
  } else {
    x[i] = v;
  }
}

__global__ void __launch_bounds__(256)
maxpool_kernel(const float* __restrict__ y, float* __restrict__ out)
{
  size_t i = (size_t)blockIdx.x * blockDim.x + threadIdx.x;
  if (i >= (size_t)B * M2 * N) return;
  const float* p = y + i * NK;
  float m = p[0];
  #pragma unroll
  for (int k = 1; k < NK; ++k) m = fmaxf(m, p[k]);
  out[i] = m;
}

// ---------------------------------------------------------------------------
// Host launch
// ---------------------------------------------------------------------------
extern "C" void kernel_launch(void* const* d_in, const int* in_sizes, int n_in,
                              void* d_out, int out_size, void* d_ws, size_t ws_size,
                              hipStream_t stream)
{
  (void)in_sizes; (void)n_in; (void)out_size; (void)ws_size;
  const float* pc1    = (const float*)d_in[0];
  const float* pc2    = (const float*)d_in[1];
  const float* feat1  = (const float*)d_in[2];
  const float* feat2  = (const float*)d_in[3];
  const float* knn1   = (const float*)d_in[4];
  const float* knn2   = (const float*)d_in[5];
  const float* t11_w  = (const float*)d_in[6];
  const float* t11_b  = (const float*)d_in[7];
  const float* t22_w  = (const float*)d_in[8];
  const float* t22_b  = (const float*)d_in[9];
  const float* pos_w  = (const float*)d_in[10];
  const float* pos_b  = (const float*)d_in[11];
  const float* gn0_w  = (const float*)d_in[12];
  const float* gn0_b  = (const float*)d_in[13];
  const float* conv1_w = (const float*)d_in[14];
  const float* conv1_b = (const float*)d_in[15];
  const float* gn1_w  = (const float*)d_in[16];
  const float* gn1_b  = (const float*)d_in[17];
  const float* conv2_w = (const float*)d_in[18];
  const float* conv2_b = (const float*)d_in[19];
  const float* gn2_w  = (const float*)d_in[20];
  const float* gn2_b  = (const float*)d_in[21];
  float* out = (float*)d_out;

  const int NM = N * NK;   // 65536
  const int C2 = C / 2;

  // ---- workspace carve-up ----
  char* ws = (char*)d_ws;
  size_t off = 0;
  auto take = [&](size_t bytes) { char* p = ws + off; off += (bytes + 255) & ~(size_t)255; return p; };
  float*    p_a    = (float*)   take((size_t)B * M0 * N * 4);
  float*    p_b    = (float*)   take((size_t)B * M0 * N * 4);
  unsigned* fpk1   = (unsigned*)take((size_t)B * C2 * N * 4);
  unsigned* fpk2   = (unsigned*)take((size_t)B * C2 * N * 4);
  unsigned* anpk1  = (unsigned*)take((size_t)B * C2 * N * 4);
  unsigned* anpk2  = (unsigned*)take((size_t)B * C2 * N * 4);
  int*      kidx   = (int*)     take((size_t)B * N * NK * 4);
  unsigned* t11pk  = (unsigned*)take((size_t)M0 * C2 * 4);
  unsigned* t22pk  = (unsigned*)take((size_t)M0 * C2 * 4);
  unsigned* c1pk   = (unsigned*)take((size_t)M0 * C2 * 4);
  unsigned* c2pk   = (unsigned*)take((size_t)M2 * C2 * 4);
  float*    x0     = (float*)   take((size_t)B * M0 * NM * 4);
  unsigned* x0pk   = (unsigned*)take((size_t)B * (M0 / 2) * NM * 4);
  float*    h1     = (float*)   take((size_t)B * M0 * NM * 4);
  unsigned* h1pk   = (unsigned*)take((size_t)B * (M0 / 2) * NM * 4);
  float*    y2     = (float*)   take((size_t)B * M2 * NM * 4);
  float*    stats  = (float*)   take(128 * 4);
  const size_t STATS_BYTES = 128 * 4;

  const float invcnt = 1.0f / (16.0f * (float)NM);   // GN count = 16ch * N * K

  // ---- one-time packing (bf16 operand prep) ----
  pack_w_kernel<<<dim3((M0 * C2 + 255) / 256), dim3(256), 0, stream>>>(t11_w, t11pk, M0, C);
  pack_w_kernel<<<dim3((M0 * C2 + 255) / 256), dim3(256), 0, stream>>>(t22_w, t22pk, M0, C);
  pack_w_kernel<<<dim3((M0 * C2 + 255) / 256), dim3(256), 0, stream>>>(conv1_w, c1pk, M0, C);
  pack_w_kernel<<<dim3((M2 * C2 + 255) / 256), dim3(256), 0, stream>>>(conv2_w, c2pk, M2, C);
  pack_x_kernel<<<dim3((unsigned)(((size_t)B * C2 * N + 255) / 256)), dim3(256), 0, stream>>>(feat1, fpk1, C, N);
  pack_x_kernel<<<dim3((unsigned)(((size_t)B * C2 * N + 255) / 256)), dim3(256), 0, stream>>>(feat2, fpk2, C, N);
  normalize_pk<<<dim3((B * N + 255) / 256), dim3(256), 0, stream>>>(knn1, anpk1);
  normalize_pk<<<dim3((B * N + 255) / 256), dim3(256), 0, stream>>>(knn2, anpk2);

  for (int d = 0; d < 2; ++d) {
    const unsigned* fA = (d == 0) ? fpk1 : fpk2;
    const unsigned* fB = (d == 0) ? fpk2 : fpk1;
    const unsigned* aA = (d == 0) ? anpk1 : anpk2;
    const unsigned* aB = (d == 0) ? anpk2 : anpk1;
    const float*    xA = (d == 0) ? pc1 : pc2;
    const float*    xB = (d == 0) ? pc2 : pc1;

    // p_a = t11_w @ fA + t11_b ; p_b = t22_w @ fB + t22_b
    gemm_wmma_bf16pk<<<dim3(N / 32, M0 / 32, B), dim3(128), 0, stream>>>(
        t11pk, fA, t11_b, p_a, M0, C, N);
    gemm_wmma_bf16pk<<<dim3(N / 32, M0 / 32, B), dim3(128), 0, stream>>>(
        t22pk, fB, t22_b, p_b, M0, C, N);

    // neighbor indices (8 euclid + 8 cosine)
    topk_kernel<<<dim3(N / 16, B), dim3(32), 0, stream>>>(aA, aB, xA, xB, kidx);

    // stage0 + GN0
    hipMemsetAsync(stats, 0, STATS_BYTES, stream);
    stage0_kernel<<<dim3(N / 16, M0 / 16, B), dim3(256), 0, stream>>>(
        p_a, p_b, kidx, xA, xB, pos_w, pos_b, x0, stats);
    gn_apply_leaky<<<dim3((unsigned)(((size_t)B * M0 * NM + 255) / 256)), dim3(256), 0, stream>>>(
        x0, stats, gn0_w, gn0_b, (unsigned short*)x0pk, M0, NM, 16, invcnt);

    // conv1 + GN1
    gemm_wmma_bf16pk<<<dim3(NM / 32, M0 / 32, B), dim3(128), 0, stream>>>(
        c1pk, x0pk, conv1_b, h1, M0, M0, NM);
    hipMemsetAsync(stats, 0, STATS_BYTES, stream);
    gn_stats_kernel<<<dim3(16, 16, B), dim3(256), 0, stream>>>(h1, stats, M0, NM, 16);
    gn_apply_leaky<<<dim3((unsigned)(((size_t)B * M0 * NM + 255) / 256)), dim3(256), 0, stream>>>(
        h1, stats, gn1_w, gn1_b, (unsigned short*)h1pk, M0, NM, 16, invcnt);

    // conv2 + GN2
    gemm_wmma_bf16pk<<<dim3(NM / 32, M2 / 32, B), dim3(128), 0, stream>>>(
        c2pk, h1pk, conv2_b, y2, M2, M0, NM);
    hipMemsetAsync(stats, 0, STATS_BYTES, stream);
    gn_stats_kernel<<<dim3(16, 32, B), dim3(256), 0, stream>>>(y2, stats, M2, NM, 32);
    gn_apply_leaky<<<dim3((unsigned)(((size_t)B * M2 * NM + 255) / 256)), dim3(256), 0, stream>>>(
        y2, stats, gn2_w, gn2_b, (unsigned short*)nullptr, M2, NM, 32, invcnt);

    // max over K -> output half d
    maxpool_kernel<<<dim3((unsigned)(((size_t)B * M2 * N + 255) / 256)), dim3(256), 0, stream>>>(
        y2, out + (size_t)d * B * M2 * N);
  }
}
